// OwnLSTM_41695542510360
// MI455X (gfx1250) — matrix operand order repaired and checked by reading, not compiled, not run on hardware
//
#include <hip/hip_runtime.h>
#include <hip/hip_bf16.h>

// CDNA5 (gfx1250) wave32 WMMA LSTM, v3.
// Changes vs v2:
//  * A-operand strips (x rows / h rows) are staged into LDS once per block with
//    GLOBAL_LOAD_ASYNC_TO_LDS_B128 (ASYNCcnt-tracked, inline asm) and read back
//    with ds_load_b128 — 8x reuse across the block's waves.
//  * xproj: block = 8 waves = 1 M-tile x 8 N-tiles. step: block = 8 waves =
//    1 M-tile x 8 J-tiles (all four gates per wave).

typedef __attribute__((ext_vector_type(16))) __bf16 v16bf;
typedef __attribute__((ext_vector_type(8)))  __bf16 v8bf;
typedef __attribute__((ext_vector_type(8)))  float  v8f;

#define HH 512
#define SS 256
#define UU 256
#define NGATES 2048   // 4*H

static __device__ __forceinline__ float sigmoid_fast(float x) {
    return 1.0f / (1.0f + __expf(-x));
}

// ---- CDNA5 async global->LDS copy (16 bytes per lane), ASYNCcnt tracked ----
static __device__ __forceinline__ unsigned lds_offset_of(const void* p) {
    return (unsigned)(uintptr_t)(const __attribute__((address_space(3))) void*)p;
}
static __device__ __forceinline__ void async_copy_b128(const void* lds_dst, const void* gsrc) {
    unsigned lds = lds_offset_of(lds_dst);
    unsigned long long ga = (unsigned long long)(uintptr_t)gsrc;
    asm volatile("global_load_async_to_lds_b128 %0, %1, off"
                 :: "v"(lds), "v"(ga) : "memory");
}
static __device__ __forceinline__ void wait_async_zero() {
    asm volatile("s_wait_asynccnt 0x0" ::: "memory");
}

// -------------------- fp32 -> bf16 bulk conversion (run once per operand) ----
__global__ void __launch_bounds__(256)
cvt_f32_bf16_kernel(const float* __restrict__ src, __bf16* __restrict__ dst, int n) {
    int i = (blockIdx.x * blockDim.x + threadIdx.x) * 8;   // n is a multiple of 8
    if (i >= n) return;
    float4 a = *(const float4*)(src + i);
    float4 b = *(const float4*)(src + i + 4);
    v8bf o;
    o[0] = (__bf16)a.x; o[1] = (__bf16)a.y; o[2] = (__bf16)a.z; o[3] = (__bf16)a.w;
    o[4] = (__bf16)b.x; o[5] = (__bf16)b.y; o[6] = (__bf16)b.z; o[7] = (__bf16)b.w;
    *(v8bf*)(dst + i) = o;
}

// ---- WMMA fragment loaders (contiguous k = k0 + 16*half + s, same order in A and B) ----

// A fragment from an LDS-resident 16x512 strip (row-major, ld = HH).
// lane<16: row lane, K = k0..k0+15; lane>=16: row lane-16, K = k0+16..k0+31.
static __device__ __forceinline__ v16bf load_a_lds(const __bf16* __restrict__ As,
                                                   int k0, int lane) {
    const int half = lane >> 4, lr = lane & 15;
    const __bf16* p = As + lr * HH + k0 + half * 16;
    v8bf lo = *(const v8bf*)p;
    v8bf hi = *(const v8bf*)(p + 8);
    v16bf a;
#pragma unroll
    for (int e = 0; e < 8; ++e) { a[e] = lo[e]; a[8 + e] = hi[e]; }
    return a;
}

// B fragment: 32x16 (KxN) with B[k][n] = W[n][k] (W row-major, ld=H):
// one contiguous 32-byte (16 x bf16) load per lane.
static __device__ __forceinline__ v16bf load_b_bf(const __bf16* __restrict__ wrow,
                                                  int k0, int half) {
    return *(const v16bf*)(wrow + k0 + half * 16);
}

// Stage a contiguous 16-row x 512-col bf16 strip (16 KB) into LDS: 256 threads
// x 4 async b128 transfers each.
static __device__ __forceinline__ void stage_strip(__bf16* As, const __bf16* gsrc) {
#pragma unroll
    for (int c = 0; c < 4; ++c) {
        const int e = (threadIdx.x + c * 256) * 8;   // element index, 8 bf16 = 16 B
        async_copy_b128(As + e, gsrc + e);
    }
    wait_async_zero();
    __syncthreads();
}

// -------------------- Kernel 1: x projection for all timesteps --------------------
// G[s*U+u, g*H+j] = x[s,u,:] @ Wg^T + bg   (bias folded into C init), fp32 out.
// Block: 256 threads = 8 waves = one M-tile x 8 consecutive N-tiles.
__global__ void __launch_bounds__(256)
lstm_xproj_kernel(const __bf16* __restrict__ xbf,                     // [S*U, H] bf16
                  const __bf16* __restrict__ Wi, const __bf16* __restrict__ Wf,
                  const __bf16* __restrict__ Wo, const __bf16* __restrict__ Wc,
                  const float* __restrict__ bi, const float* __restrict__ bf,
                  const float* __restrict__ bo, const float* __restrict__ bc,
                  float* __restrict__ G)                              // [S*U, 4H] fp32
{
    __shared__ __bf16 As[16 * HH];                 // 16 KB staged A strip

    const int lane = threadIdx.x & 31;
    const int w    = threadIdx.x >> 5;             // wave in block: 0..7
    const int NTG  = (NGATES / 16) / 8;            // 16 N-tile groups
    const int mt   = blockIdx.x / NTG;
    const int ntg  = blockIdx.x % NTG;
    const int m0   = mt * 16;
    const int n0   = (ntg * 8 + w) * 16;

    stage_strip(As, xbf + (size_t)m0 * HH);        // async -> LDS, then barrier

    const int g = n0 / HH;                         // gate id, uniform per wave
    const int nloc0 = n0 - g * HH;
    const __bf16* Wg = (g == 0) ? Wi : (g == 1) ? Wf : (g == 2) ? Wo : Wc;
    const float*  bg = (g == 0) ? bi : (g == 1) ? bf : (g == 2) ? bo : bc;

    const int half = lane >> 4, lr = lane & 15;
    const __bf16* wrow = Wg + (size_t)(nloc0 + lr) * HH;

    const float bias = bg[nloc0 + lr];
    v8f acc;
#pragma unroll
    for (int r = 0; r < 8; ++r) acc[r] = bias;

#pragma unroll 4
    for (int k0 = 0; k0 < HH; k0 += 32) {
        v16bf a = load_a_lds(As, k0, lane);
        v16bf b = load_b_bf(wrow, k0, half);
        acc = __builtin_amdgcn_wmma_f32_16x16x32_bf16(false, a, false, b,
                                                      (short)0, acc, false, false);
    }

    // C/D layout: element r -> row m0 + half*8 + r, col n0 + lr
#pragma unroll
    for (int r = 0; r < 8; ++r)
        G[(size_t)(m0 + half * 8 + r) * NGATES + n0 + lr] = acc[r];
}

// -------------------- Kernel 2: one recurrent timestep --------------------
// Block: 256 threads = 8 waves = one M-tile (16 users) x 8 J-tiles; each wave
// computes all 4 gate tiles for its patch (shared LDS A fragment, 4 WMMAs/K-step).
__global__ void __launch_bounds__(256)
lstm_step_kernel(const __bf16* __restrict__ hprev,  // [U, H] bf16 (ping buffer)
                 const float*  __restrict__ Gt,     // [U, 4H] precomputed x-part (+bias)
                 const __bf16* __restrict__ Ui, const __bf16* __restrict__ Uf,
                 const __bf16* __restrict__ Uo, const __bf16* __restrict__ Uc,
                 float*  __restrict__ cstate,       // [U, H] in/out (wave-exclusive patch)
                 float*  __restrict__ hout,         // [U, H] fp32 == out[t]
                 __bf16* __restrict__ houtbf)       // [U, H] bf16 (pong buffer)
{
    __shared__ __bf16 As[16 * HH];                 // 16 KB staged h strip

    const int lane = threadIdx.x & 31;
    const int w    = threadIdx.x >> 5;             // wave in block: 0..7
    const int JTG  = (HH / 16) / 8;                // 4 J-tile groups
    const int mt   = blockIdx.x / JTG;
    const int jtg  = blockIdx.x % JTG;
    const int m0   = mt * 16;
    const int j0   = (jtg * 8 + w) * 16;
    const int half = lane >> 4, lr = lane & 15;

    stage_strip(As, hprev + (size_t)m0 * HH);      // async -> LDS, then barrier

    const __bf16* uirow = Ui + (size_t)(j0 + lr) * HH;
    const __bf16* ufrow = Uf + (size_t)(j0 + lr) * HH;
    const __bf16* uorow = Uo + (size_t)(j0 + lr) * HH;
    const __bf16* ucrow = Uc + (size_t)(j0 + lr) * HH;

    // Init accumulators with precomputed x-projection (+bias) for each gate.
    v8f ai, af_, ao, ac;
#pragma unroll
    for (int r = 0; r < 8; ++r) {
        const size_t row = (size_t)(m0 + half * 8 + r) * NGATES + j0 + lr;
        ai[r]  = Gt[row + 0 * HH];
        af_[r] = Gt[row + 1 * HH];
        ao[r]  = Gt[row + 2 * HH];
        ac[r]  = Gt[row + 3 * HH];
    }

    // Recurrent GEMM: LDS A fragment feeds 4 WMMAs per K-step.
#pragma unroll 2
    for (int k0 = 0; k0 < HH; k0 += 32) {
        v16bf a  = load_a_lds(As, k0, lane);
        v16bf b0 = load_b_bf(uirow, k0, half);
        v16bf b1 = load_b_bf(ufrow, k0, half);
        v16bf b2 = load_b_bf(uorow, k0, half);
        v16bf b3 = load_b_bf(ucrow, k0, half);
        ai  = __builtin_amdgcn_wmma_f32_16x16x32_bf16(false, a, false, b0, (short)0, ai,  false, false);
        af_ = __builtin_amdgcn_wmma_f32_16x16x32_bf16(false, a, false, b1, (short)0, af_, false, false);
        ao  = __builtin_amdgcn_wmma_f32_16x16x32_bf16(false, a, false, b2, (short)0, ao,  false, false);
        ac  = __builtin_amdgcn_wmma_f32_16x16x32_bf16(false, a, false, b3, (short)0, ac,  false, false);
    }

    // Gate nonlinearity + state update in registers; this wave exclusively owns
    // its (u, j) patch of cstate, so the in-place c update is race-free.
#pragma unroll
    for (int r = 0; r < 8; ++r) {
        const int u = m0 + half * 8 + r;
        const int j = j0 + lr;
        const size_t idx = (size_t)u * HH + j;
        const float ig = sigmoid_fast(ai[r]);
        const float fg = sigmoid_fast(af_[r]);
        const float og = sigmoid_fast(ao[r]);
        const float ct = tanhf(ac[r]);
        const float cn = fg * cstate[idx] + ig * ct;
        const float hn = og * tanhf(cn);
        cstate[idx]  = cn;
        hout[idx]    = hn;
        houtbf[idx]  = (__bf16)hn;
    }
}

extern "C" void kernel_launch(void* const* d_in, const int* in_sizes, int n_in,
                              void* d_out, int out_size, void* d_ws, size_t ws_size,
                              hipStream_t stream) {
    const float* x  = (const float*)d_in[0];
    const float* h0 = (const float*)d_in[1];
    const float* c0 = (const float*)d_in[2];
    const float* Wf32[8] = { (const float*)d_in[3], (const float*)d_in[4],
                             (const float*)d_in[5], (const float*)d_in[6],
                             (const float*)d_in[7], (const float*)d_in[8],
                             (const float*)d_in[9], (const float*)d_in[10] };
    const float* bi = (const float*)d_in[11];
    const float* bf = (const float*)d_in[12];
    const float* bo = (const float*)d_in[13];
    const float* bc = (const float*)d_in[14];

    float* out = (float*)d_out;            // [S*U*H] out ++ [U*H] h_f ++ [U*H] c_f

    // ---- workspace carve-up (all chunks comfortably aligned) ----
    char* ws = (char*)d_ws;
    float*  c_ws  = (float*)ws;                          ws += (size_t)UU * HH * 4;   // cell state
    __bf16* hbuf0 = (__bf16*)ws;                         ws += (size_t)UU * HH * 2;   // h ping
    __bf16* hbuf1 = (__bf16*)ws;                         ws += (size_t)UU * HH * 2;   // h pong
    __bf16* wbf[8];
    for (int i = 0; i < 8; ++i) { wbf[i] = (__bf16*)ws;  ws += (size_t)HH * HH * 2; } // Wi..Wc,Ui..Uc
    __bf16* xbf   = (__bf16*)ws;                         ws += (size_t)SS * UU * HH * 2;
    float*  G     = (float*)ws;                          // [S*U, 4H] fp32

    // c state <- c0
    hipMemcpyAsync(c_ws, c0, sizeof(float) * UU * HH, hipMemcpyDeviceToDevice, stream);

    // ---- one-time fp32 -> bf16 conversions ----
    auto cvt = [&](const float* src, __bf16* dst, int n) {
        cvt_f32_bf16_kernel<<<dim3((n / 8 + 255) / 256), dim3(256), 0, stream>>>(src, dst, n);
    };
    for (int i = 0; i < 8; ++i) cvt(Wf32[i], wbf[i], HH * HH);
    cvt(h0, hbuf0, UU * HH);
    cvt(x, xbf, SS * UU * HH);

    // ---- time-parallel input projection (+bias folded in) ----
    {
        const int blocks = (SS * UU / 16) * ((NGATES / 16) / 8);   // 4096 * 16 = 65536
        lstm_xproj_kernel<<<dim3(blocks), dim3(256), 0, stream>>>(
            xbf, wbf[0], wbf[1], wbf[2], wbf[3], bi, bf, bo, bc, G);
    }

    // ---- sequential recurrence: bf16 h ping-pongs, fp32 h streams to d_out ----
    __bf16* hping[2] = { hbuf0, hbuf1 };
    for (int t = 0; t < SS; ++t) {
        const __bf16* hprev = hping[t & 1];
        __bf16* hnext = hping[(t + 1) & 1];
        const float* Gt = G + (size_t)t * UU * NGATES;
        const int blocks = (UU / 16) * ((HH / 16) / 8);            // 16 * 4 = 64
        lstm_step_kernel<<<dim3(blocks), dim3(256), 0, stream>>>(
            hprev, Gt, wbf[4], wbf[5], wbf[6], wbf[7],
            c_ws, out + (size_t)t * UU * HH, hnext);
    }

    // ---- tail outputs: h_f = out[S-1], c_f = running c state ----
    hipMemcpyAsync(out + (size_t)SS * UU * HH,
                   out + (size_t)(SS - 1) * UU * HH,
                   sizeof(float) * UU * HH, hipMemcpyDeviceToDevice, stream);
    hipMemcpyAsync(out + (size_t)SS * UU * HH + (size_t)UU * HH,
                   c_ws,
                   sizeof(float) * UU * HH, hipMemcpyDeviceToDevice, stream);
}
